// myLSTM_69818988364208
// MI455X (gfx1250) — compile-verified, matrix operand order checked by reading, not run on hardware
//
#include <hip/hip_runtime.h>
#include <hip/hip_bf16.h>

typedef __attribute__((ext_vector_type(16))) __bf16 v16bf;
typedef __attribute__((ext_vector_type(8)))  __bf16 v8bf;
typedef __attribute__((ext_vector_type(8)))  float  v8f;

#define LSTM_B 64
#define LSTM_S 1024
#define LSTM_D 512
#define LSTM_H 512
#define NGATE  2048   // 4 gates * H
#define NWG    32     // persistent workgroups in the scan

// ---- helpers ---------------------------------------------------------------

__device__ __forceinline__ __bf16 f2bf(float f) {
  unsigned u = __builtin_bit_cast(unsigned, f);
  unsigned r = u + 0x7FFFu + ((u >> 16) & 1u);       // round-to-nearest-even
  return __builtin_bit_cast(__bf16, (unsigned short)(r >> 16));
}

// Load one 16x32 bf16 A-fragment from a row-major [M][ld] bf16 matrix.
// Wave32 layout: lane holds row = (lane&15); halves 0..7 -> K = kb..kb+7,
// halves 8..15 -> K = kb+16..kb+23, with kb = kt*32 + (lane>>4)*8.
__device__ __forceinline__ v16bf load_a16x32(const __bf16* base, int row, int ld, int kb) {
  const __bf16* p = base + (size_t)row * ld + kb;
  v8bf lo = *(const v8bf*)(p);
  v8bf hi = *(const v8bf*)(p + 16);
  return __builtin_shufflevector(lo, hi, 0,1,2,3,4,5,6,7,8,9,10,11,12,13,14,15);
}

__device__ __forceinline__ float sigmoidf(float x) { return 1.f / (1.f + __expf(-x)); }

// ---- prep kernels ----------------------------------------------------------

__global__ void lstm_init(const float* __restrict__ bf_, const float* __restrict__ bi_,
                          const float* __restrict__ bc_, const float* __restrict__ bo_,
                          float* __restrict__ bias4, __bf16* __restrict__ hA,
                          __bf16* __restrict__ hB, unsigned* __restrict__ bar) {
  int i = blockIdx.x * 256 + threadIdx.x;
  if (i < LSTM_H) {
    bias4[i]              = bf_[i];
    bias4[LSTM_H   + i]   = bi_[i];
    bias4[2*LSTM_H + i]   = bc_[i];
    bias4[3*LSTM_H + i]   = bo_[i];
  }
  if (i < LSTM_B * LSTM_H) { hA[i] = f2bf(0.f); hB[i] = f2bf(0.f); }
  if (i == 0) *bar = 0u;
}

// x (fp32, [B*S][D]) -> bf16 row-major
__global__ void lstm_cvt_x(const float* __restrict__ x, __bf16* __restrict__ xb) {
  size_t i = ((size_t)blockIdx.x * 256 + threadIdx.x) * 4;
  float4 v = *(const float4*)(x + i);
  xb[i+0] = f2bf(v.x); xb[i+1] = f2bf(v.y); xb[i+2] = f2bf(v.z); xb[i+3] = f2bf(v.w);
}

// Pack B-fragments for both GEMMs.  nt (0..127) spans 4 gates * 32 N-tiles;
// dst index = ((nt*16 + kt)*32 + lane)*16 + half; element = W_g[row][n] with
//   n = (nt&31)*16 + (lane&15)
//   k = kt*32 + ((half&8)?16:0) + (lane>>4)*8 + (half&7)
//   row = k (recurrent "bh") or 512+k (input "bx")
__global__ void lstm_pack(const float* __restrict__ Wf, const float* __restrict__ Wi,
                          const float* __restrict__ Wc, const float* __restrict__ Wo,
                          __bf16* __restrict__ bx, __bf16* __restrict__ bh) {
  unsigned i = blockIdx.x * 256 + threadIdx.x;     // 0 .. 2*2^20-1
  unsigned which = i >> 20;                        // 0 = bx (input), 1 = bh (recurrent)
  unsigned r = i & 0xFFFFFu;
  unsigned half = r & 15u;
  unsigned lane = (r >> 4) & 31u;
  unsigned kt   = (r >> 9) & 15u;
  unsigned nt   = r >> 13;                         // 0..127
  unsigned gate = nt >> 5;
  unsigned n    = (nt & 31u) * 16u + (lane & 15u);
  unsigned klocal = ((half & 8u) ? 16u : 0u) + ((lane >> 4) << 3) + (half & 7u);
  unsigned k = kt * 32u + klocal;
  const float* W = (gate == 0) ? Wf : (gate == 1) ? Wi : (gate == 2) ? Wc : Wo;
  unsigned row = which ? k : (LSTM_H + k);
  float v = W[(size_t)row * LSTM_H + n];
  (which ? bh : bx)[r] = f2bf(v);
}

// ---- phase 1: xproj[t][b][gatecol] = x[b,t,:] @ Wx + bias ------------------
// M = B*S = 65536, N = 2048, K = 512.  2M x 2N register tile per wave:
// 4 accumulators, each A/B fragment reused twice -> 1 load per WMMA.

__device__ __forceinline__ void xproj_store_tile(float* __restrict__ xproj,
                                                 const float* __restrict__ bias4,
                                                 v8f acc, int mtile, int nt, int lane) {
  int n = nt * 16 + (lane & 15);
  float bias = bias4[n];
  int rbase = mtile * 16 + ((lane >> 4) << 3);
#pragma unroll
  for (int r = 0; r < 8; ++r) {
    int m = rbase + r;
    int b = m >> 10;               // / S
    int t = m & 1023;              // % S
    xproj[((size_t)t * LSTM_B + b) * NGATE + n] = acc[r] + bias;
  }
}

__global__ void lstm_xproj(const __bf16* __restrict__ xb, const __bf16* __restrict__ bx,
                           const float* __restrict__ bias4, float* __restrict__ xproj) {
  int wid  = blockIdx.x * 8 + (threadIdx.x >> 5);  // 0..131071
  int lane = threadIdx.x & 31;
  int mt2 = wid >> 6;              // 0..2047  (pairs of M-tiles)
  int nt2 = wid & 63;              // 0..63    (pairs of N-tiles)
  int arow0 = (mt2 * 2) * 16 + (lane & 15);
  int arow1 = arow0 + 16;
  int kb0   = (lane >> 4) * 8;
  v8f acc00 = {}, acc01 = {}, acc10 = {}, acc11 = {};
#pragma unroll
  for (int kt = 0; kt < 16; ++kt) {
    int kb = kt * 32 + kb0;
    v16bf a0 = load_a16x32(xb, arow0, LSTM_D, kb);
    v16bf a1 = load_a16x32(xb, arow1, LSTM_D, kb);
    v16bf b0 = *(const v16bf*)(bx + ((((size_t)(nt2 * 2 + 0) * 16 + kt) * 32 + lane) << 4));
    v16bf b1 = *(const v16bf*)(bx + ((((size_t)(nt2 * 2 + 1) * 16 + kt) * 32 + lane) << 4));
    acc00 = __builtin_amdgcn_wmma_f32_16x16x32_bf16(false, a0, false, b0, (short)0, acc00, false, false);
    acc01 = __builtin_amdgcn_wmma_f32_16x16x32_bf16(false, a0, false, b1, (short)0, acc01, false, false);
    acc10 = __builtin_amdgcn_wmma_f32_16x16x32_bf16(false, a1, false, b0, (short)0, acc10, false, false);
    acc11 = __builtin_amdgcn_wmma_f32_16x16x32_bf16(false, a1, false, b1, (short)0, acc11, false, false);
  }
  xproj_store_tile(xproj, bias4, acc00, mt2 * 2 + 0, nt2 * 2 + 0, lane);
  xproj_store_tile(xproj, bias4, acc01, mt2 * 2 + 0, nt2 * 2 + 1, lane);
  xproj_store_tile(xproj, bias4, acc10, mt2 * 2 + 1, nt2 * 2 + 0, lane);
  xproj_store_tile(xproj, bias4, acc11, mt2 * 2 + 1, nt2 * 2 + 1, lane);
}

// ---- phase 2: persistent recurrent scan ------------------------------------
// 32 blocks * 256 threads. Block w owns h columns [16w,16w+16) and the four
// matching gate column tiles. Wave handles one M-tile and two gates, sharing
// the h A-fragment across both gate WMMAs. Gates staged via LDS; c in regs.
__global__ void __launch_bounds__(256, 1)
lstm_scan(const __bf16* __restrict__ bh, const float* __restrict__ xproj,
          __bf16* __restrict__ hA, __bf16* __restrict__ hB, unsigned* __restrict__ bar,
          float* __restrict__ out_hidden, float* __restrict__ out_h,
          float* __restrict__ out_c) {
  __shared__ float stag[4 * LSTM_B * 16];   // gates[g][b][j], 16 KB
  const int tid  = threadIdx.x;
  const int w    = blockIdx.x;              // 0..31
  const int lane = tid & 31;
  const int wave = tid >> 5;
  const int kb0  = (lane >> 4) * 8;
  const int mtile = wave >> 1;              // batch tile 0..3
  const int g0    = (wave & 1) * 2;         // this wave computes gates g0, g0+1
  const int ntg0  = g0 * 32 + w;
  const int ntg1  = (g0 + 1) * 32 + w;
  const int arow  = mtile * 16 + (lane & 15);

  float creg[4];                            // this thread's 4 c-state cells
#pragma unroll
  for (int q = 0; q < 4; ++q) creg[q] = 0.f;

  for (int t = 0; t < LSTM_S; ++t) {
    const __bf16* hprev = (t & 1) ? hB : hA;
    __bf16*       hnext = (t & 1) ? hA : hB;

    // Prefetch this step's xproj slice (one lane per 16-thread group per line)
    if ((tid & 15) == 0) {
#pragma unroll
      for (int q = 0; q < 4; ++q) {
        int b = (tid + q * 256) >> 4;
        const float* p = xproj + ((size_t)t * LSTM_B + b) * NGATE + w * 16;
        __builtin_prefetch(p + 0 * LSTM_H, 0, 1);
        __builtin_prefetch(p + 1 * LSTM_H, 0, 1);
        __builtin_prefetch(p + 2 * LSTM_H, 0, 1);
        __builtin_prefetch(p + 3 * LSTM_H, 0, 1);
      }
    }

    v8f acc0 = {}, acc1 = {};
#pragma unroll
    for (int kt = 0; kt < 16; ++kt) {
      v16bf a  = load_a16x32(hprev, arow, LSTM_H, kt * 32 + kb0);
      v16bf b0 = *(const v16bf*)(bh + ((((size_t)ntg0 * 16 + kt) * 32 + lane) << 4));
      v16bf b1 = *(const v16bf*)(bh + ((((size_t)ntg1 * 16 + kt) * 32 + lane) << 4));
      acc0 = __builtin_amdgcn_wmma_f32_16x16x32_bf16(false, a, false, b0, (short)0, acc0, false, false);
      acc1 = __builtin_amdgcn_wmma_f32_16x16x32_bf16(false, a, false, b1, (short)0, acc1, false, false);
    }
    {
      int j = lane & 15;
      int rbase = mtile * 16 + ((lane >> 4) << 3);
#pragma unroll
      for (int r = 0; r < 8; ++r) {
        stag[(g0 + 0) * (LSTM_B * 16) + (rbase + r) * 16 + j] = acc0[r];
        stag[(g0 + 1) * (LSTM_B * 16) + (rbase + r) * 16 + j] = acc1[r];
      }
    }
    __syncthreads();

#pragma unroll
    for (int q = 0; q < 4; ++q) {
      int idx = tid + q * 256;              // b*16 + j
      int b = idx >> 4;
      int j = idx & 15;
      int jg = w * 16 + j;
      size_t xo = ((size_t)t * LSTM_B + b) * NGATE;
      float af = stag[0 * 1024 + idx] + xproj[xo + 0 * LSTM_H + jg];
      float ai = stag[1 * 1024 + idx] + xproj[xo + 1 * LSTM_H + jg];
      float ag = stag[2 * 1024 + idx] + xproj[xo + 2 * LSTM_H + jg];
      float ao = stag[3 * 1024 + idx] + xproj[xo + 3 * LSTM_H + jg];
      float fg = sigmoidf(af);
      float ig = sigmoidf(ai);
      float gg = tanhf(ag);
      float og = sigmoidf(ao);
      float c  = creg[q] * fg + gg * ig;
      float h  = og * tanhf(c);
      creg[q] = c;
      out_hidden[((size_t)b * LSTM_S + t) * LSTM_H + jg] = h;
      hnext[b * LSTM_H + jg] = f2bf(h);
      if (t == LSTM_S - 1) {
        out_h[b * LSTM_H + jg] = h;
        out_c[b * LSTM_H + jg] = c;
      }
    }
    __syncthreads();

    // grid-wide barrier: monotonic counter, 1024*32 < 2^32
    if (tid == 0) {
      __threadfence();
      atomicAdd(bar, 1u);
      unsigned target = (unsigned)(t + 1) * NWG;
      volatile unsigned* vb = (volatile unsigned*)bar;
      while (*vb < target) { __builtin_amdgcn_s_sleep(1); }
      __threadfence();
    }
    __syncthreads();
  }
}

// ---- host glue -------------------------------------------------------------

static inline size_t al256(size_t x) { return (x + 255) & ~(size_t)255; }

extern "C" void kernel_launch(void* const* d_in, const int* in_sizes, int n_in,
                              void* d_out, int out_size, void* d_ws, size_t ws_size,
                              hipStream_t stream) {
  const float* x  = (const float*)d_in[0];
  const float* Wf = (const float*)d_in[1];
  const float* bf = (const float*)d_in[2];
  const float* Wi = (const float*)d_in[3];
  const float* bi = (const float*)d_in[4];
  const float* Wc = (const float*)d_in[5];
  const float* bc = (const float*)d_in[6];
  const float* Wo = (const float*)d_in[7];
  const float* bo = (const float*)d_in[8];

  char* ws = (char*)d_ws;
  size_t off = 0;
  unsigned* bar = (unsigned*)(ws + off); off = al256(off + 256);
  float*  bias4 = (float*)(ws + off);    off = al256(off + (size_t)NGATE * 4);
  __bf16* bx    = (__bf16*)(ws + off);   off = al256(off + (size_t)128*16*32*16 * 2);
  __bf16* bh    = (__bf16*)(ws + off);   off = al256(off + (size_t)128*16*32*16 * 2);
  __bf16* xb    = (__bf16*)(ws + off);   off = al256(off + (size_t)LSTM_B*LSTM_S*LSTM_D * 2);
  __bf16* hA    = (__bf16*)(ws + off);   off = al256(off + (size_t)LSTM_B*LSTM_H * 2);
  __bf16* hB    = (__bf16*)(ws + off);   off = al256(off + (size_t)LSTM_B*LSTM_H * 2);
  float*  xproj = (float*)(ws + off);    off = al256(off + (size_t)LSTM_S*LSTM_B*NGATE * 4);

  float* out_hidden = (float*)d_out;
  float* out_h = out_hidden + (size_t)LSTM_B * LSTM_S * LSTM_H;
  float* out_c = out_h + (size_t)LSTM_B * LSTM_H;

  lstm_init<<<128, 256, 0, stream>>>(bf, bi, bc, bo, bias4, hA, hB, bar);
  lstm_cvt_x<<<(LSTM_B * LSTM_S * LSTM_D) / (256 * 4), 256, 0, stream>>>(x, xb);
  lstm_pack<<<8192, 256, 0, stream>>>(Wf, Wi, Wc, Wo, bx, bh);
  lstm_xproj<<<16384, 256, 0, stream>>>(xb, bx, bias4, xproj);
  lstm_scan<<<NWG, 256, 0, stream>>>(bh, xproj, hA, hB, bar, out_hidden, out_h, out_c);
}
